// GreedyDecoder_11879879544139
// MI455X (gfx1250) — matrix operand order repaired and checked by reading, not compiled, not run on hardware
//
#include <hip/hip_runtime.h>
#include <hip/hip_bf16.h>

#define H     1024
#define V     50257
#define LENC  128
#define STEPS 32
#define STOPID 658

#define NB   128
#define BLK  256
#define NTHREADS (NB * BLK)
#define NWAVES   (NTHREADS / 32)

typedef __attribute__((ext_vector_type(16))) __bf16       v16bf;
typedef __attribute__((ext_vector_type(8)))  float        v8f;
typedef __attribute__((ext_vector_type(4)))  unsigned int uint4v;

struct WS {
    unsigned int bar_cnt;
    unsigned int bar_gen;
    int tok;
    int done;
    unsigned long long best;     // packed argmax: key<<32 | (0xFFFFFFFF - vocab)
    unsigned int pad0[58];       // pad header to 256 B
    float h[H];
    float h_new[H];
    float emb[H];
    float att[LENC];
    float aw[LENC];
    float ctx[H];
    float x[H];
    unsigned short hb[H];        // bf16(h_new) for WMMA A operand
};

__device__ inline float wave_sum(float v) {
    #pragma unroll
    for (int off = 16; off; off >>= 1) v += __shfl_xor(v, off, 32);
    return v;
}

__device__ inline unsigned long long wave_max_u64(unsigned long long v) {
    #pragma unroll
    for (int off = 16; off; off >>= 1) {
        unsigned hi = __shfl_xor((unsigned)(v >> 32), off, 32);
        unsigned lo = __shfl_xor((unsigned)(v & 0xFFFFFFFFull), off, 32);
        unsigned long long o = ((unsigned long long)hi << 32) | lo;
        v = (o > v) ? o : v;
    }
    return v;
}

__device__ inline unsigned short f32_to_bf16_rne(float f) {
    unsigned u = __float_as_uint(f);
    return (unsigned short)((u + 0x7FFFu + ((u >> 16) & 1u)) >> 16);
}

// Global-atomic grid barrier (all NB blocks must be co-resident; tiny LDS/VGPR
// footprint of this kernel guarantees that for 128 x 256-thread workgroups).
__device__ inline void grid_sync(WS* ws) {
    __syncthreads();
    if (threadIdx.x == 0) {
        __threadfence();  // release
        volatile unsigned* gen = (volatile unsigned*)&ws->bar_gen;
        unsigned g = *gen;
        if (atomicAdd(&ws->bar_cnt, 1u) == (unsigned)(NB - 1)) {
            ws->bar_cnt = 0;
            __threadfence();
            atomicAdd(&ws->bar_gen, 1u);
        } else {
            while (*gen == g) { __builtin_amdgcn_s_sleep(1); }
        }
        __threadfence();  // acquire
    }
    __syncthreads();
}

__global__ void cvt_bf16_kernel(const float* __restrict__ src,
                                unsigned short* __restrict__ dst, long n) {
    long i = (long)blockIdx.x * blockDim.x + threadIdx.x;
    long stride = (long)gridDim.x * blockDim.x;
    for (; i < n; i += stride) dst[i] = f32_to_bf16_rne(src[i]);
}

__global__ void init_kernel(const int* __restrict__ dec_in,
                            const float* __restrict__ h0, WS* ws) {
    int t = blockIdx.x * blockDim.x + threadIdx.x;
    if (t == 0) {
        ws->bar_cnt = 0; ws->bar_gen = 0;
        ws->tok = dec_in[0]; ws->done = 0; ws->best = 0ull;
    }
    for (int j = t; j < H; j += blockDim.x * gridDim.x) ws->h[j] = h0[j];
}

__global__ __launch_bounds__(BLK) void decode_kernel(
    const float* __restrict__ enc,     const float* __restrict__ emb_tab,
    const float* __restrict__ attn_w,  const float* __restrict__ attn_b,
    const float* __restrict__ comb_w,  const float* __restrict__ comb_b,
    const float* __restrict__ wih,     const float* __restrict__ whh,
    const float* __restrict__ bih,     const float* __restrict__ bhh,
    const float* __restrict__ out_w,   const float* __restrict__ out_b,
    float* __restrict__ out_ids, WS* ws, const unsigned short* __restrict__ owb)
{
    const int tid  = blockIdx.x * BLK + threadIdx.x;
    const int lane = threadIdx.x & 31;
    const int wave = tid >> 5;

    for (int s = 0; s < STEPS; ++s) {
        // ---- Early stop: done is stable since the last step-final barrier and
        // identical across all blocks -> uniform break. Remaining ids are 0. ----
        {
            int dn = ((volatile int*)&ws->done)[0];
            if (dn) {
                if (tid == 0)
                    for (int r = s; r < STEPS; ++r) out_ids[r] = 0.f;
                break;
            }
        }

        // ---- P1: embedding gather + attention logits [128 x 2048 GEMV] ----
        int tok = ((volatile int*)&ws->tok)[0];
        const float* erow = emb_tab + (size_t)tok * H;
        for (int j = tid; j < H; j += NTHREADS) ws->emb[j] = erow[j];
        for (int r = wave; r < LENC; r += NWAVES) {
            const float* wr = attn_w + (size_t)r * (2 * H);
            float acc = 0.f;
            for (int k = lane; k < H; k += 32) acc += erow[k] * wr[k];
            for (int k = lane; k < H; k += 32) acc += ws->h[k] * wr[H + k];
            acc = wave_sum(acc);
            if (lane == 0) ws->att[r] = acc + attn_b[r];
        }
        grid_sync(ws);

        // ---- P2: softmax over 128 (wave 0 only) ----
        if (wave == 0) {
            float v[4]; float m = -3.4e38f;
            #pragma unroll
            for (int j = 0; j < 4; ++j) { v[j] = ws->att[lane * 4 + j]; m = fmaxf(m, v[j]); }
            #pragma unroll
            for (int off = 16; off; off >>= 1) m = fmaxf(m, __shfl_xor(m, off, 32));
            float sm = 0.f;
            #pragma unroll
            for (int j = 0; j < 4; ++j) { v[j] = expf(v[j] - m); sm += v[j]; }
            sm = wave_sum(sm);
            float inv = 1.f / sm;
            #pragma unroll
            for (int j = 0; j < 4; ++j) ws->aw[lane * 4 + j] = v[j] * inv;
        }
        grid_sync(ws);

        // ---- P3: ctx = aw @ encoder_outputs ----
        for (int j = tid; j < H; j += NTHREADS) {
            float acc = 0.f;
            for (int i = 0; i < LENC; ++i) acc += ws->aw[i] * enc[(size_t)i * H + j];
            ws->ctx[j] = acc;
        }
        grid_sync(ws);

        // ---- P4: x = relu(concat(emb,ctx) @ comb_w.T + b) [1024 x 2048] ----
        for (int r = wave; r < H; r += NWAVES) {
            const float* wr = comb_w + (size_t)r * (2 * H);
            float acc = 0.f;
            for (int k = lane; k < H; k += 32) acc += ws->emb[k] * wr[k];
            for (int k = lane; k < H; k += 32) acc += ws->ctx[k] * wr[H + k];
            acc = wave_sum(acc);
            if (lane == 0) ws->x[r] = fmaxf(acc + comb_b[r], 0.f);
        }
        grid_sync(ws);

        // ---- P5: fused GRU step (wave per output element, 6 dots each) ----
        for (int i = wave; i < H; i += NWAVES) {
            const float* a0 = wih + (size_t)i * H;
            const float* a1 = wih + (size_t)(i + H) * H;
            const float* a2 = wih + (size_t)(i + 2 * H) * H;
            const float* b0 = whh + (size_t)i * H;
            const float* b1 = whh + (size_t)(i + H) * H;
            const float* b2 = whh + (size_t)(i + 2 * H) * H;
            float ir = 0.f, iz = 0.f, in_ = 0.f, hr = 0.f, hz = 0.f, hn = 0.f;
            for (int k = lane; k < H; k += 32) {
                float xv = ws->x[k], hv = ws->h[k];
                ir += xv * a0[k]; iz += xv * a1[k]; in_ += xv * a2[k];
                hr += hv * b0[k]; hz += hv * b1[k]; hn += hv * b2[k];
            }
            ir = wave_sum(ir); iz = wave_sum(iz); in_ = wave_sum(in_);
            hr = wave_sum(hr); hz = wave_sum(hz); hn = wave_sum(hn);
            if (lane == 0) {
                float rg = 1.f / (1.f + expf(-(ir + bih[i]        + hr + bhh[i])));
                float zg = 1.f / (1.f + expf(-(iz + bih[H + i]    + hz + bhh[H + i])));
                float ng = tanhf(in_ + bih[2 * H + i] + rg * (hn + bhh[2 * H + i]));
                float hnew = (1.f - zg) * ng + zg * ws->h[i];
                ws->h_new[i] = hnew;
                ws->hb[i]    = f32_to_bf16_rne(hnew);
            }
        }
        if (tid == 0) ws->best = 0ull;
        grid_sync(ws);

        // ---- P6: logits = h_new @ out_w.T  + global argmax ----
        unsigned long long mybest = 0ull;
        if (owb) {
            // bf16 path: v_wmma_f32_16x16x32_bf16, 16 vocab rows per wave-tile.
            const int T    = (V + 15) / 16;
            const int half = lane >> 4;     // K half selector
            const int ln   = lane & 15;     // N (vocab within tile)
            const unsigned keep = (ln == 0) ? 0xFFFFFFFFu : 0u;  // A row 0 only
            for (int t = wave; t < T; t += NWAVES) {
                int base = t * 16;
                int row  = base + ln; if (row >= V) row = V - 1;
                const unsigned short* bp = owb + (size_t)row * H;
                v8f c = {};
                for (int kb = 0; kb < H; kb += 32) {
                    // A: lane0 -> K[kb..kb+7],K[kb+16..23]; lane16 -> +8 within halves
                    uint4v alo = *(const uint4v*)(ws->hb + kb + half * 8);
                    uint4v ahi = *(const uint4v*)(ws->hb + kb + 16 + half * 8);
                    alo &= keep; ahi &= keep;
                    // B: lane holds K[half*8 .. +7], K[16+half*8 .. +7] of its vocab row
                    uint4v blo = *(const uint4v*)(bp + kb + half * 8);
                    uint4v bhi = *(const uint4v*)(bp + kb + 16 + half * 8);
                    __builtin_prefetch(bp + kb + 64, 0, 1);
                    union { uint4v u[2]; v16bf v; } ua, ub;
                    ua.u[0] = alo; ua.u[1] = ahi;
                    ub.u[0] = blo; ub.u[1] = bhi;
                    c = __builtin_amdgcn_wmma_f32_16x16x32_bf16(
                            false, ua.v, false, ub.v, (short)0, c, false, false);
                }
                int vidx  = base + ln;
                int bidx  = vidx < V ? vidx : V - 1;
                float lg  = c[0] + out_b[bidx];   // C row M=0 lives in lanes 0..15, VGPR0
                unsigned u = __float_as_uint(lg);
                u = (u & 0x80000000u) ? ~u : (u | 0x80000000u);
                unsigned long long key =
                    ((unsigned long long)u << 32) | (unsigned)(0xFFFFFFFFu - (unsigned)vidx);
                key = (half == 0 && vidx < V) ? key : 0ull;
                mybest = (key > mybest) ? key : mybest;
            }
            mybest = wave_max_u64(mybest);
        } else {
            // fp32 fallback (workspace too small for bf16 copy): wave per row.
            for (int r = wave; r < V; r += NWAVES) {
                const float* wr = out_w + (size_t)r * H;
                float acc = 0.f;
                for (int k = lane; k < H; k += 32) acc += ws->h_new[k] * wr[k];
                acc = wave_sum(acc);
                float lg = acc + out_b[r];
                unsigned u = __float_as_uint(lg);
                u = (u & 0x80000000u) ? ~u : (u | 0x80000000u);
                unsigned long long key =
                    ((unsigned long long)u << 32) | (unsigned)(0xFFFFFFFFu - (unsigned)r);
                mybest = (key > mybest) ? key : mybest;
            }
        }
        if (lane == 0 && mybest) atomicMax(&ws->best, mybest);
        grid_sync(ws);

        // ---- P7: pick token, emit id, advance state ----
        {
            unsigned long long best = ((volatile unsigned long long*)&ws->best)[0];
            int top = (int)(0xFFFFFFFFu - (unsigned)(best & 0xFFFFFFFFull));
            int old_done = ((volatile int*)&ws->done)[0];
            // h_next = done ? h : h_new   (race with tid0's done-write is benign:
            // it only matters the step the stop fires, after which h is unused)
            if (!old_done)
                for (int j = tid; j < H; j += NTHREADS) ws->h[j] = ws->h_new[j];
            if (tid == 0) {
                out_ids[s] = (float)(old_done ? 0 : top);
                ws->tok  = top;
                ws->done = old_done | (top == 1) | (top == STOPID);
            }
        }
        grid_sync(ws);
    }
}

extern "C" void kernel_launch(void* const* d_in, const int* in_sizes, int n_in,
                              void* d_out, int out_size, void* d_ws, size_t ws_size,
                              hipStream_t stream) {
    const int*   dec_in = (const int*)  d_in[0];
    const float* h0     = (const float*)d_in[1];
    const float* enc    = (const float*)d_in[2];
    const float* embt   = (const float*)d_in[3];
    const float* attn_w = (const float*)d_in[4];
    const float* attn_b = (const float*)d_in[5];
    const float* comb_w = (const float*)d_in[6];
    const float* comb_b = (const float*)d_in[7];
    const float* wih    = (const float*)d_in[8];
    const float* whh    = (const float*)d_in[9];
    const float* bih    = (const float*)d_in[10];
    const float* bhh    = (const float*)d_in[11];
    const float* out_w  = (const float*)d_in[12];
    const float* out_b  = (const float*)d_in[13];

    WS* ws = (WS*)d_ws;
    size_t hdr  = (sizeof(WS) + 255) & ~(size_t)255;
    size_t need = hdr + (size_t)V * H * sizeof(unsigned short);
    unsigned short* owb = nullptr;
    if (ws_size >= need) {
        owb = (unsigned short*)((char*)d_ws + hdr);
        // Re-done every call (deterministic); ~310 MB traffic ≈ 15 us at HBM peak.
        cvt_bf16_kernel<<<2048, 256, 0, stream>>>(out_w, owb, (long)V * H);
    }
    init_kernel<<<4, 256, 0, stream>>>(dec_in, h0, ws);
    decode_kernel<<<NB, BLK, 0, stream>>>(enc, embt, attn_w, attn_b, comb_w, comb_b,
                                          wih, whh, bih, bhh, out_w, out_b,
                                          (float*)d_out, ws, owb);
}
